// VQCodebookEMA_43499428774156
// MI455X (gfx1250) — matrix operand order repaired and checked by reading, not compiled, not run on hardware
//
#include <hip/hip_runtime.h>
#include <math.h>

// ---------------------------------------------------------------------------
// VQ-VAE codebook EMA step for MI455X (gfx1250, wave32, WMMA).
// Distance search (z @ E^T) as bf16 WMMA w/ f32 accumulation.
// argmin_n(x2 - 2 z.e + e2) == argmin_n(e2 - 2 z.e), so x2 is skipped.
// e-chunks staged via GLOBAL_LOAD_ASYNC_TO_LDS_B128 (double-buffered, ASYNCcnt);
// A/B fragments software-pipelined in registers.
// ---------------------------------------------------------------------------

typedef __attribute__((ext_vector_type(16))) __bf16 v16bf;
typedef __attribute__((ext_vector_type(8)))  float  v8f;

// pointer types for the async-to-LDS builtin: (v4i AS1*, v4i AS3*, imm, imm)
typedef int vq_v4i __attribute__((ext_vector_type(4)));
typedef __attribute__((address_space(1))) vq_v4i* vq_gp1;
typedef __attribute__((address_space(3))) vq_v4i* vq_lp3;

#if __has_builtin(__builtin_amdgcn_global_load_async_to_lds_b128)
#define VQ_ASYNC_LDS 1
#else
#define VQ_ASYNC_LDS 0
#endif

#if __has_builtin(__builtin_amdgcn_s_wait_asynccnt)
#define VQ_WAIT_ASYNC(n) __builtin_amdgcn_s_wait_asynccnt(n)
#else
#define VQ_WAIT_ASYNC(n) asm volatile("s_wait_asynccnt %0" ::"i"(n) : "memory")
#endif

namespace vqk {
constexpr int   Dd = 512;
constexpr int   Kk = 4096;
constexpr int   Mm = 64 * 512;          // B*S tokens
constexpr float kBeta  = 0.25f;
constexpr float kDecay = 0.99f;
constexpr float kEps   = 1e-5f;
constexpr int   TM = 128;               // token rows per block
constexpr int   TN = 64;                // codes per LDS chunk (double-buffered)
constexpr int   NSUB = TN / 16;         // 4 N-subtiles per chunk
constexpr int   NCH  = Kk / TN;         // 64 chunks
constexpr int   LROW = Dd + 8;          // padded LDS row (bf16): conflict-free b128
constexpr int   CHUNK_ITEMS = TN * (Dd / 8);       // 4096 uint4 per chunk
constexpr int   PER_THR = CHUNK_ITEMS / 256;       // 16 async b128 per thread
}
using namespace vqk;

// ---- convert z (fp32) -> bf16, 8 elems per thread --------------------------
__global__ __launch_bounds__(256) void k_cvt_z(const float* __restrict__ z,
                                               __bf16* __restrict__ zb) {
  size_t i = ((size_t)blockIdx.x * 256 + threadIdx.x) * 8;
  float4 a = *(const float4*)(z + i);
  float4 b = *(const float4*)(z + i + 4);
  union { __bf16 h[8]; uint4 q; } u;
  u.h[0] = (__bf16)a.x; u.h[1] = (__bf16)a.y; u.h[2] = (__bf16)a.z; u.h[3] = (__bf16)a.w;
  u.h[4] = (__bf16)b.x; u.h[5] = (__bf16)b.y; u.h[6] = (__bf16)b.z; u.h[7] = (__bf16)b.w;
  *(uint4*)(zb + i) = u.q;
}

// ---- convert embeddings -> bf16 and compute e2[k] --------------------------
__global__ __launch_bounds__(256) void k_cvt_e(const float* __restrict__ emb,
                                               __bf16* __restrict__ eb,
                                               float* __restrict__ e2) {
  __shared__ float red[256];
  const int k = blockIdx.x, t = threadIdx.x;
  float ss = 0.f;
  for (int j = t; j < Dd; j += 256) {
    float v = emb[(size_t)k * Dd + j];
    eb[(size_t)k * Dd + j] = (__bf16)v;
    ss += v * v;
  }
  red[t] = ss; __syncthreads();
  for (int o = 128; o > 0; o >>= 1) { if (t < o) red[t] += red[t + o]; __syncthreads(); }
  if (t == 0) e2[k] = red[0];
}

// ---- chunk staging: global -> LDS -----------------------------------------
__device__ __forceinline__ void issue_chunk(const __bf16* __restrict__ eb, int nb,
                                            __bf16* __restrict__ dst, int tid) {
#pragma unroll
  for (int t = 0; t < PER_THR; ++t) {
    int i = tid + t * 256;
    int row = i >> 6, c = (i & 63) << 3;
#if VQ_ASYNC_LDS
    __builtin_amdgcn_global_load_async_to_lds_b128(
        (vq_gp1)(uintptr_t)(eb + (size_t)(nb + row) * Dd + c),
        (vq_lp3)(uintptr_t)(dst + row * LROW + c),
        0, 0);
#else
    *(uint4*)&dst[row * LROW + c] = *(const uint4*)(eb + (size_t)(nb + row) * Dd + c);
#endif
  }
}

// ---- main: WMMA distance GEMM + running argmin -----------------------------
// 256 threads = 8 waves; wave owns a 16-row strip; chunk = 64 codes (4 subtiles).
__global__ __launch_bounds__(256) void k_argmin(const __bf16* __restrict__ zb,
                                                const __bf16* __restrict__ eb,
                                                const float* __restrict__ e2,
                                                int* __restrict__ idxo) {
  extern __shared__ char smem[];
  __bf16* zt = (__bf16*)smem;                            // [TM][LROW]
  __bf16* et0 = (__bf16*)(smem + (size_t)TM * LROW * 2); // [TN][LROW] buf 0
  __bf16* et1 = et0 + (size_t)TN * LROW;                 // [TN][LROW] buf 1

  const int tid  = threadIdx.x;
  const int wave = tid >> 5;
  const int lane = tid & 31;
  const int l16  = lane & 15;
  const int hh   = lane >> 4;            // wave32 half
  const int mbase = blockIdx.x * TM;

#if VQ_ASYNC_LDS
  issue_chunk(eb, 0, et0, tid);          // chunk 0 in flight
  issue_chunk(eb, TN, et1, tid);         // chunk 1 in flight
#endif

  // stage z tile (synchronous; covered by first loop barrier)
  for (int i = tid; i < TM * (Dd / 8); i += 256) {
    int row = i >> 6, c = (i & 63) << 3;
    *(uint4*)&zt[row * LROW + c] = *(const uint4*)(zb + (size_t)(mbase + row) * Dd + c);
  }

  float bestD[8];
  int   bestI[8];
#pragma unroll
  for (int r = 0; r < 8; ++r) { bestD[r] = 3.4e38f; bestI[r] = 0; }

  const int arow = (wave << 4) + l16;    // A-fragment source row (M = lane%16)

  union frag { v16bf v; uint4 q[2]; };

  for (int c = 0; c < NCH; ++c) {
    __bf16* buf = (c & 1) ? et1 : et0;
#if VQ_ASYNC_LDS
    // drain own async loads for chunk c (chunk c+1 may stay in flight)
    if (c + 1 < NCH) { VQ_WAIT_ASYNC(PER_THR); } else { VQ_WAIT_ASYNC(0); }
    __syncthreads();                      // all threads' chunk c visible
#else
    __syncthreads();                      // prior compute done reading buf
    issue_chunk(eb, c * TN, buf, tid);    // synchronous copy fallback
    __syncthreads();
#endif

    v8f acc[NSUB];
    const v8f vzero = {0.f, 0.f, 0.f, 0.f, 0.f, 0.f, 0.f, 0.f};
#pragma unroll
    for (int nt = 0; nt < NSUB; ++nt) acc[nt] = vzero;

    // software-pipelined fragments: prefetch step s+1 while issuing WMMAs of s
    frag A[2];
    frag Bf[2][NSUB];
    {
      const __bf16* ap = &zt[arow * LROW + (hh << 3)];
      A[0].q[0] = *(const uint4*)ap;
      A[0].q[1] = *(const uint4*)(ap + 16);
#pragma unroll
      for (int nt = 0; nt < NSUB; ++nt) {
        const __bf16* bp = &buf[((nt << 4) + l16) * LROW + (hh << 3)];
        Bf[0][nt].q[0] = *(const uint4*)bp;
        Bf[0][nt].q[1] = *(const uint4*)(bp + 16);
      }
    }
#pragma unroll
    for (int s = 0; s < Dd / 32; ++s) {
      const int cu = s & 1, nx = cu ^ 1;
      if (s + 1 < Dd / 32) {
        // A 16x32 bf16: elems 0..7 -> K=h*8+[0..7]; elems 8..15 -> K=16+h*8+[0..7]
        const __bf16* ap = &zt[arow * LROW + (s + 1) * 32 + (hh << 3)];
        A[nx].q[0] = *(const uint4*)ap;
        A[nx].q[1] = *(const uint4*)(ap + 16);
#pragma unroll
        for (int nt = 0; nt < NSUB; ++nt) {
          // B 32x16 bf16: N = lane%16, elems 0..7 -> K=h*8+[0..7]; elems 8..15 -> K=16+h*8+[0..7]
          const __bf16* bp = &buf[((nt << 4) + l16) * LROW + (s + 1) * 32 + (hh << 3)];
          Bf[nx][nt].q[0] = *(const uint4*)bp;
          Bf[nx][nt].q[1] = *(const uint4*)(bp + 16);
        }
      }
#pragma unroll
      for (int nt = 0; nt < NSUB; ++nt) {
        acc[nt] = __builtin_amdgcn_wmma_f32_16x16x32_bf16(
            false, A[cu].v, false, Bf[cu][nt].v, (short)0, acc[nt], false, false);
      }
    }

    // dist = e2[n] - 2*dot ; running per-lane argmin (n increases -> strict <)
#pragma unroll
    for (int nt = 0; nt < NSUB; ++nt) {
      int   n  = c * TN + (nt << 4) + l16;
      float ev = e2[n];
#pragma unroll
      for (int r = 0; r < 8; ++r) {
        float dist = ev - 2.0f * acc[nt][r];
        if (dist < bestD[r]) { bestD[r] = dist; bestI[r] = n; }
      }
    }

#if VQ_ASYNC_LDS
    __syncthreads();                      // everyone done reading buf
    if (c + 2 < NCH) issue_chunk(eb, (c + 2) * TN, buf, tid);
#endif
  }

  // reduce across the 16 lanes of each half (xor masks 8..1 stay in-half)
#pragma unroll
  for (int r = 0; r < 8; ++r) {
    float d = bestD[r]; int i = bestI[r];
#pragma unroll
    for (int off = 8; off >= 1; off >>= 1) {
      float d2 = __shfl_xor(d, off, 32);
      int   i2 = __shfl_xor(i, off, 32);
      if (d2 < d || (d2 == d && i2 < i)) { d = d2; i = i2; }
    }
    if (l16 == 0) idxo[mbase + (wave << 4) + (hh << 3) + r] = i;  // M = r + 8*half
  }
}

// ---- quantize + masked-MSE partials + segment sums -------------------------
__global__ __launch_bounds__(256) void k_quant(const float* __restrict__ z,
                                               const unsigned char* __restrict__ mask,
                                               const float* __restrict__ emb,
                                               const int* __restrict__ idx,
                                               float* __restrict__ o_zq,
                                               float* __restrict__ o_idx,
                                               float* __restrict__ sums,
                                               float* __restrict__ counts,
                                               float* __restrict__ scal) {
  __shared__ float red[256];
  const int m = blockIdx.x, t = threadIdx.x;
  const int ix = idx[m];
  const bool mv = mask[m] != 0;
  float acc = 0.f;
  for (int j = t; j < Dd; j += 256) {
    float zv = z[(size_t)m * Dd + j];
    float ev = emb[(size_t)ix * Dd + j];
    o_zq[(size_t)m * Dd + j] = mv ? ev : zv;   // straight-through == e_q where valid
    if (mv) {
      float df = zv - ev;
      acc += df * df;
      atomicAdd(&sums[(size_t)ix * Dd + j], zv);
    }
  }
  red[t] = acc; __syncthreads();
  for (int o = 128; o > 0; o >>= 1) { if (t < o) red[t] += red[t + o]; __syncthreads(); }
  if (t == 0) {
    if (mv) { atomicAdd(&scal[0], red[0]); atomicAdd(&scal[1], 1.f); atomicAdd(&counts[ix], 1.f); }
    o_idx[m] = mv ? (float)ix : -1.f;
  }
}

// ---- EMA counts + total ----------------------------------------------------
__global__ __launch_bounds__(256) void k_counts(const float* __restrict__ ema_counts,
                                                const float* __restrict__ counts,
                                                float* __restrict__ o_cnt,
                                                float* __restrict__ scal) {
  int k = blockIdx.x * 256 + threadIdx.x;
  float nc = ema_counts[k] * kDecay + counts[k] * (1.f - kDecay);
  o_cnt[k] = nc;
  atomicAdd(&scal[2], nc);
}

// ---- EMA sums + normalized embeddings --------------------------------------
__global__ __launch_bounds__(256) void k_sums(const float* __restrict__ ema_sums,
                                              const float* __restrict__ sums,
                                              const float* __restrict__ o_cnt,
                                              float* __restrict__ o_sum,
                                              float* __restrict__ o_emb) {
  size_t i = (size_t)blockIdx.x * 256 + threadIdx.x;
  int k = (int)(i / Dd);
  float ns = ema_sums[i] * kDecay + sums[i] * (1.f - kDecay);
  o_sum[i] = ns;
  o_emb[i] = ns / (o_cnt[k] + kEps);
}

// ---- entropy partial -------------------------------------------------------
__global__ __launch_bounds__(256) void k_entropy(const float* __restrict__ o_cnt,
                                                 float* __restrict__ scal) {
  int k = blockIdx.x * 256 + threadIdx.x;
  float tot = scal[2];
  float p = o_cnt[k] / (tot + kEps);
  p = fmaxf(p, kEps);
  atomicAdd(&scal[3], p * logf(p + kEps));
}

// ---- finalize scalars ------------------------------------------------------
__global__ void k_final(const float* __restrict__ scal,
                        float* __restrict__ o_vq, float* __restrict__ o_ppl) {
  float mse = scal[0] / (fmaxf(scal[1], 1.f) * (float)Dd);
  *o_vq  = (1.f + kBeta) * mse;
  *o_ppl = expf(-scal[3]);
}

// ---------------------------------------------------------------------------
extern "C" void kernel_launch(void* const* d_in, const int* in_sizes, int n_in,
                              void* d_out, int out_size, void* d_ws, size_t ws_size,
                              hipStream_t stream) {
  (void)in_sizes; (void)n_in; (void)out_size; (void)ws_size;
  const float*         z    = (const float*)d_in[0];
  const unsigned char* mask = (const unsigned char*)d_in[1];  // numpy bool = 1 byte
  const float*         emb  = (const float*)d_in[2];
  const float*         emac = (const float*)d_in[3];
  const float*         emas = (const float*)d_in[4];
  float* out = (float*)d_out;

  // workspace carve-up
  char* ws = (char*)d_ws;
  size_t off = 0;
  __bf16* zb  = (__bf16*)(ws + off); off += (size_t)Mm * Dd * 2;
  __bf16* eb  = (__bf16*)(ws + off); off += (size_t)Kk * Dd * 2;
  float*  e2  = (float*)(ws + off);  off += (size_t)Kk * 4;
  int*    idx = (int*)(ws + off);    off += (size_t)Mm * 4;
  size_t zstart = off;
  float*  cnt  = (float*)(ws + off); off += (size_t)Kk * 4;
  float*  sums = (float*)(ws + off); off += (size_t)Kk * Dd * 4;
  float*  scal = (float*)(ws + off); off += 4 * 4;
  (void)hipMemsetAsync(ws + zstart, 0, off - zstart, stream);  // zero accumulators

  // output carve-up (tuple concatenated flat, fp32)
  float* o_zq  = out;                          // [M,D]
  float* o_idx = o_zq  + (size_t)Mm * Dd;      // [M]
  float* o_vq  = o_idx + Mm;                   // [1]
  float* o_emb = o_vq  + 1;                    // [K,D]
  float* o_cnt = o_emb + (size_t)Kk * Dd;      // [K]
  float* o_sum = o_cnt + Kk;                   // [K,D]
  float* o_ppl = o_sum + (size_t)Kk * Dd;      // [1]

  k_cvt_z<<<(Mm * Dd) / (256 * 8), 256, 0, stream>>>(z, zb);
  k_cvt_e<<<Kk, 256, 0, stream>>>(emb, eb, e2);

  size_t smem = (size_t)(TM + 2 * TN) * LROW * 2;  // ~260 KB dynamic LDS
  k_argmin<<<Mm / TM, 256, smem, stream>>>(zb, eb, e2, idx);

  k_quant<<<Mm, 256, 0, stream>>>(z, mask, emb, idx, o_zq, o_idx, sums, cnt, scal);
  k_counts<<<Kk / 256, 256, 0, stream>>>(emac, cnt, o_cnt, scal);
  k_sums<<<(Kk * Dd) / 256, 256, 0, stream>>>(emas, sums, o_cnt, o_sum, o_emb);
  k_entropy<<<Kk / 256, 256, 0, stream>>>(o_cnt, scal);
  k_final<<<1, 1, 0, stream>>>(scal, o_vq, o_ppl);
}
